// RWKV_53961969106959
// MI455X (gfx1250) — matrix operand order repaired and checked by reading, not compiled, not run on hardware
//
#include <hip/hip_runtime.h>

// ---------------------------------------------------------------------------
// RWKV-7 forward for MI455X (gfx1250, wave32).  All large GEMMs run through
// v_wmma_f32_16x16x32_bf16 (f32->bf16 RNE in LDS, f32 accumulate), 128x128
// block tiles, vectorized b128 staging, prefetch of streamed weights.
// Sequential time-mix scan, LayerNorms and elementwise glue are VALU kernels.
// ---------------------------------------------------------------------------

#define TT 1024      // sequence length
#define DD 1024      // model dim
#define HH 16        // heads
#define KKH 64       // head dim

typedef __attribute__((ext_vector_type(16))) __bf16 v16bf;
typedef __attribute__((ext_vector_type(8)))  float  v8f;
typedef __attribute__((ext_vector_type(4)))  unsigned int u32x4;
typedef __attribute__((ext_vector_type(2)))  unsigned int u32x2;
typedef __attribute__((ext_vector_type(4)))  float  f32x4;

__device__ __forceinline__ unsigned short f32_to_bf16(float f) {
  union { float f; unsigned int u; } c; c.f = f;
  unsigned int u = c.u;
  unsigned int r = u + 0x7FFFu + ((u >> 16) & 1u);   // round-to-nearest-even
  return (unsigned short)(r >> 16);
}
__device__ __forceinline__ unsigned int pack_bf16x2(float lo, float hi) {
  return (unsigned int)f32_to_bf16(lo) | ((unsigned int)f32_to_bf16(hi) << 16);
}
__device__ __forceinline__ float sigmoidf_(float x) { return 1.f / (1.f + expf(-x)); }

// ---------------------------------------------------------------------------
// GEMM: C[M,N] = act(A[M,K] @ B[K,N]); fp32 row-major operands.
// Requirements (all call sites satisfy): M % 128 == 0, K % 32 == 0, N % 32 == 0.
// Block tile 128x128, K-step 32, 8 wave32/block, each wave a 32x64 strip
// (2 A-frags x 4 B-frags -> 8 WMMA per K-step).
// act: 0=none, 1=relu^2, 2=sigmoid, 3=tanh.  ntst: nontemporal C stores.
// ---------------------------------------------------------------------------
#define GTM 128
#define GTN 128
#define GKS 32
#define LDS_STR 40   // padded stride (halves): breaks LDS bank conflicts

__global__ __launch_bounds__(256) void gemm_bf16_wmma(
    const float* __restrict__ A, const float* __restrict__ B,
    float* __restrict__ C, int M, int N, int K, int act, int ntst)
{
  __shared__ __attribute__((aligned(16))) unsigned short Alds[GTM * LDS_STR];
  __shared__ __attribute__((aligned(16))) unsigned short Blds[GTN * LDS_STR];
  const int tid  = threadIdx.x;
  const int row0 = blockIdx.y * GTM;
  const int col0 = blockIdx.x * GTN;
  const int wave = tid >> 5;
  const int lane = tid & 31;
  const int wm = wave & 3;     // M strip: 32 rows at wm*32
  const int wn = wave >> 2;    // N strip: 64 cols at wn*64
  const int g  = lane >> 4;    // lane group (ISA 16x16 fragment layout)
  const int lr = lane & 15;

  v8f z = {0.f,0.f,0.f,0.f,0.f,0.f,0.f,0.f};
  v8f acc[2][4];
#pragma unroll
  for (int mt = 0; mt < 2; ++mt)
#pragma unroll
    for (int nt = 0; nt < 4; ++nt) acc[mt][nt] = z;

  // per-thread B staging coordinates (reused for prefetch)
  const int pk2  = tid >> 5;          // k-pair index 0..7 for it=0
  const int pnn4 = (tid & 31) << 2;   // n offset 0..124

  for (int k0 = 0; k0 < K; k0 += GKS) {
    // prefetch next K-step of the streamed B operand into GL2
    if (k0 + GKS < K && col0 + pnn4 < N)
      __builtin_prefetch(B + (size_t)(k0 + GKS + (pk2 << 1)) * N + col0 + pnn4, 0, 1);

    // ---- stage A tile (128 x 32) as bf16, row-major [m][k]; no guards ----
#pragma unroll
    for (int it = 0; it < (GTM * GKS / 4) / 256; ++it) {   // 4 iters
      int idx4 = it * 256 + tid;
      int r  = idx4 >> 3;           // 0..127
      int c4 = (idx4 & 7) << 2;     // 0,4,...,28
      f32x4 av = *(const f32x4*)(A + (size_t)(row0 + r) * K + (k0 + c4));
      u32x2 pk;
      pk.x = pack_bf16x2(av.x, av.y);
      pk.y = pack_bf16x2(av.z, av.w);
      *(u32x2*)(&Alds[r * LDS_STR + c4]) = pk;
    }
    // ---- stage B tile (32 x 128) transposed -> [n][k], pair-packed ----
#pragma unroll
    for (int it = 0; it < (GKS * GTN / 8) / 256; ++it) {   // 2 iters
      int p   = it * 256 + tid;
      int k2  = p >> 5;             // 0..15
      int nn4 = (p & 31) << 2;      // 0..124
      int kk  = k2 << 1;
      int gn  = col0 + nn4;
      f32x4 b0 = {0.f,0.f,0.f,0.f}, b1 = {0.f,0.f,0.f,0.f};
      if (gn < N) {                 // N % 4 == 0 -> whole float4 in/out
        b0 = *(const f32x4*)(B + (size_t)(k0 + kk)     * N + gn);
        b1 = *(const f32x4*)(B + (size_t)(k0 + kk + 1) * N + gn);
      }
      *(unsigned int*)&Blds[(nn4 + 0) * LDS_STR + kk] = pack_bf16x2(b0.x, b1.x);
      *(unsigned int*)&Blds[(nn4 + 1) * LDS_STR + kk] = pack_bf16x2(b0.y, b1.y);
      *(unsigned int*)&Blds[(nn4 + 2) * LDS_STR + kk] = pack_bf16x2(b0.z, b1.z);
      *(unsigned int*)&Blds[(nn4 + 3) * LDS_STR + kk] = pack_bf16x2(b0.w, b1.w);
    }
    __syncthreads();

    // ---- fragments: A group g holds K = g*8+0..7 and 16+g*8+0..7 ----
    union FragU { u32x4 q[2]; v16bf v; };
    FragU fa[2];
#pragma unroll
    for (int mt = 0; mt < 2; ++mt) {
      const unsigned short* ap = &Alds[(wm * 32 + mt * 16 + lr) * LDS_STR];
      fa[mt].q[0] = *(const u32x4*)(ap + g * 8);
      fa[mt].q[1] = *(const u32x4*)(ap + 16 + g * 8);
    }
    // ---- B fragments: group g holds K = g*8+0..7 and 16+g*8+0..7 for column lr ----
    FragU fb[4];
#pragma unroll
    for (int nt = 0; nt < 4; ++nt) {
      const unsigned short* bp = &Blds[(wn * 64 + nt * 16 + lr) * LDS_STR];
      fb[nt].q[0] = *(const u32x4*)(bp + g * 8);
      fb[nt].q[1] = *(const u32x4*)(bp + 16 + g * 8);
    }
#pragma unroll
    for (int mt = 0; mt < 2; ++mt)
#pragma unroll
      for (int nt = 0; nt < 4; ++nt)
        acc[mt][nt] = __builtin_amdgcn_wmma_f32_16x16x32_bf16(
            false, fa[mt].v, false, fb[nt].v, (short)0, acc[mt][nt], false, false);
    __syncthreads();
  }

#pragma unroll
  for (int mt = 0; mt < 2; ++mt) {
#pragma unroll
    for (int nt = 0; nt < 4; ++nt) {
      int col = col0 + wn * 64 + nt * 16 + lr;
      if (col < N) {
#pragma unroll
        for (int j = 0; j < 8; ++j) {
          int row = row0 + wm * 32 + mt * 16 + g * 8 + j;   // M % 128 == 0
          float v = acc[mt][nt][j];
          if      (act == 1) { v = fmaxf(v, 0.f); v = v * v; }
          else if (act == 2) { v = sigmoidf_(v); }
          else if (act == 3) { v = tanhf(v); }
          float* cp = &C[(size_t)row * N + col];
          if (ntst) __builtin_nontemporal_store(v, cp);
          else      *cp = v;
        }
      }
    }
  }
}

// ---------------------------------------------------------------------------
// LayerNorm over D=1024, one block per row, 256 threads
// ---------------------------------------------------------------------------
__global__ __launch_bounds__(256) void ln_kernel(
    const float* __restrict__ in, const float* __restrict__ g,
    const float* __restrict__ b, float* __restrict__ out, float eps)
{
  const int t = blockIdx.x, tid = threadIdx.x;
  const float* x = in + (size_t)t * DD;
  __shared__ float s1[256], s2[256];
  float a1 = 0.f, a2 = 0.f;
  for (int d = tid; d < DD; d += 256) { float v = x[d]; a1 += v; a2 += v * v; }
  s1[tid] = a1; s2[tid] = a2; __syncthreads();
  for (int off = 128; off > 0; off >>= 1) {
    if (tid < off) { s1[tid] += s1[tid + off]; s2[tid] += s2[tid + off]; }
    __syncthreads();
  }
  float mean = s1[0] * (1.f / DD);
  float var  = s2[0] * (1.f / DD) - mean * mean;
  float rs   = rsqrtf(var + eps);
  for (int d = tid; d < DD; d += 256)
    out[(size_t)t * DD + d] = (x[d] - mean) * rs * g[d] + b[d];
}

// embedding gather + LayerNorm
__global__ __launch_bounds__(256) void embed_ln_kernel(
    const int* __restrict__ tokens, const float* __restrict__ embed,
    const float* __restrict__ g, const float* __restrict__ b,
    float* __restrict__ out)
{
  const int t = blockIdx.x, tid = threadIdx.x;
  const float* x = embed + (size_t)tokens[t] * DD;
  __shared__ float s1[256], s2[256];
  float a1 = 0.f, a2 = 0.f;
  for (int d = tid; d < DD; d += 256) { float v = x[d]; a1 += v; a2 += v * v; }
  s1[tid] = a1; s2[tid] = a2; __syncthreads();
  for (int off = 128; off > 0; off >>= 1) {
    if (tid < off) { s1[tid] += s1[tid + off]; s2[tid] += s2[tid + off]; }
    __syncthreads();
  }
  float mean = s1[0] * (1.f / DD);
  float var  = s2[0] * (1.f / DD) - mean * mean;
  float rs   = rsqrtf(var + 1e-5f);
  for (int d = tid; d < DD; d += 256)
    out[(size_t)t * DD + d] = (x[d] - mean) * rs * g[d] + b[d];
}

// token-shift with 6 lerp weights: xs_i = x + w_i*(x_prev - x)
__global__ __launch_bounds__(256) void shift6_kernel(
    const float* __restrict__ y, const float* __restrict__ ts,
    float* __restrict__ xr, float* __restrict__ xd, float* __restrict__ xk,
    float* __restrict__ xv, float* __restrict__ xi, float* __restrict__ xg)
{
  int i = blockIdx.x * 256 + threadIdx.x;
  if (i >= TT * DD) return;
  int t = i / DD, d = i - t * DD;
  float cur = y[i];
  float prev = (t == 0) ? 0.f : y[i - DD];
  float del = prev - cur;
  xr[i] = cur + ts[0 * DD + d] * del;
  xd[i] = cur + ts[1 * DD + d] * del;
  xk[i] = cur + ts[2 * DD + d] * del;
  xv[i] = cur + ts[3 * DD + d] * del;
  xi[i] = cur + ts[4 * DD + d] * del;
  xg[i] = cur + ts[5 * DD + d] * del;
}

__global__ __launch_bounds__(256) void shift1_kernel(
    const float* __restrict__ y, const float* __restrict__ w, float* __restrict__ o)
{
  int i = blockIdx.x * 256 + threadIdx.x;
  if (i >= TT * DD) return;
  int t = i / DD, d = i - t * DD;
  float cur = y[i];
  float prev = (t == 0) ? 0.f : y[i - DD];
  o[i] = cur + w[d] * (prev - cur);
}

// v = v + sigmoid(base + lo) * (v0 - v)
__global__ __launch_bounds__(256) void vmix_kernel(
    float* __restrict__ v, const float* __restrict__ v0,
    const float* __restrict__ lo, const float* __restrict__ base)
{
  int i = blockIdx.x * 256 + threadIdx.x;
  if (i >= TT * DD) return;
  int d = i & (DD - 1);
  float m = sigmoidf_(base[d] + lo[i]);
  float vv = v[i];
  v[i] = vv + m * (v0[i] - vv);
}

// decay = exp(-exp(-0.5 - softplus(-(d_base + lo))))
__global__ __launch_bounds__(256) void decay_kernel(
    const float* __restrict__ lo, const float* __restrict__ base, float* __restrict__ out)
{
  int i = blockIdx.x * 256 + threadIdx.x;
  if (i >= TT * DD) return;
  int d = i & (DD - 1);
  float z = -(base[d] + lo[i]);
  float sp = fmaxf(z, 0.f) + log1pf(expf(-fabsf(z)));   // stable softplus
  float lnd = -0.5f - sp;
  out[i] = expf(-expf(lnd));
}

// out = sigmoid(base + lo)
__global__ __launch_bounds__(256) void sigb_kernel(
    const float* __restrict__ lo, const float* __restrict__ base, float* __restrict__ out)
{
  int i = blockIdx.x * 256 + threadIdx.x;
  if (i >= TT * DD) return;
  int d = i & (DD - 1);
  out[i] = sigmoidf_(base[d] + lo[i]);
}

// per-(t,h): dk = (k*dkm) / max(||k*dkm||, 1e-12)   (uses pre-update k)
__global__ __launch_bounds__(64) void dknorm_kernel(
    const float* __restrict__ k, const float* __restrict__ dkm, float* __restrict__ dk)
{
  const int tid = threadIdx.x;
  const size_t base = (size_t)blockIdx.x * KKH;
  const int h = blockIdx.x & (HH - 1);
  float val = k[base + tid] * dkm[h * KKH + tid];
  __shared__ float sh[64];
  sh[tid] = val * val; __syncthreads();
  for (int off = 32; off > 0; off >>= 1) { if (tid < off) sh[tid] += sh[tid + off]; __syncthreads(); }
  float n = fmaxf(sqrtf(sh[0]), 1e-12f);
  dk[base + tid] = val / n;
}

// k = k + i_mix * (k*iclr - k)
__global__ __launch_bounds__(256) void kupd_kernel(
    float* __restrict__ k, const float* __restrict__ ic, const float* __restrict__ imix)
{
  int i = blockIdx.x * 256 + threadIdx.x;
  if (i >= TT * DD) return;
  int d = i & (DD - 1);
  float kk = k[i];
  k[i] = kk + imix[d] * (kk * ic[i] - kk);
}

// ---------------------------------------------------------------------------
// Sequential RWKV-7 scan: one block per head, 64 lanes = value rows of state.
// state[v][k] kept fully in registers (64 f32/lane); K-vectors staged in LDS.
// ---------------------------------------------------------------------------
__global__ __launch_bounds__(64) void scan_kernel(
    const float* __restrict__ r, const float* __restrict__ k,
    const float* __restrict__ v, const float* __restrict__ w,
    const float* __restrict__ ic, const float* __restrict__ dk,
    float* __restrict__ out)
{
  const int h = blockIdx.x;
  const int vi = threadIdx.x;
  float state[KKH];
#pragma unroll
  for (int j = 0; j < KKH; ++j) state[j] = 0.f;
  __shared__ float sr[KKH], sk[KKH], sw[KKH], sdk[KKH], sidk[KKH];

  for (int t = 0; t < TT; ++t) {
    const size_t base = ((size_t)t * HH + h) * KKH;
    float rv = r[base + vi], kv = k[base + vi], wv = w[base + vi];
    float iv = ic[base + vi], dv = dk[base + vi];
    float vt = v[base + vi];
    sr[vi] = rv; sk[vi] = kv; sw[vi] = wv; sdk[vi] = dv; sidk[vi] = iv * dv;
    __syncthreads();
    float sab = 0.f;
#pragma unroll
    for (int j = 0; j < KKH; ++j) sab += state[j] * sdk[j];
    float o = 0.f;
#pragma unroll
    for (int j = 0; j < KKH; ++j) {
      float s = state[j] * sw[j] - sab * sidk[j] + vt * sk[j];
      state[j] = s;
      o += s * sr[j];
    }
    out[base + vi] = o;
    __syncthreads();
  }
}

// per-(t,h): GroupNorm(eps=0.00064) + bonus + gate
__global__ __launch_bounds__(64) void postscan_kernel(
    const float* __restrict__ o, const float* __restrict__ r,
    const float* __restrict__ k, const float* __restrict__ v,
    const float* __restrict__ gate, const float* __restrict__ gn_g,
    const float* __restrict__ gn_b, const float* __restrict__ bonusw,
    float* __restrict__ out)
{
  const int tid = threadIdx.x;
  const size_t base = (size_t)blockIdx.x * KKH;
  const int h = blockIdx.x & (HH - 1);
  const int d = h * KKH + tid;
  float ov = o[base + tid];
  __shared__ float sh[64];
  sh[tid] = ov; __syncthreads();
  for (int off = 32; off > 0; off >>= 1) { if (tid < off) sh[tid] += sh[tid + off]; __syncthreads(); }
  float mean = sh[0] * (1.f / KKH); __syncthreads();
  float dvv = ov - mean;
  sh[tid] = dvv * dvv; __syncthreads();
  for (int off = 32; off > 0; off >>= 1) { if (tid < off) sh[tid] += sh[tid + off]; __syncthreads(); }
  float var = sh[0] * (1.f / KKH); __syncthreads();
  float gn = dvv * rsqrtf(var + 0.00064f) * gn_g[d] + gn_b[d];
  sh[tid] = r[base + tid] * k[base + tid] * bonusw[d]; __syncthreads();
  for (int off = 32; off > 0; off >>= 1) { if (tid < off) sh[tid] += sh[tid + off]; __syncthreads(); }
  float s = sh[0]; __syncthreads();
  out[base + tid] = (gn + s * v[base + tid]) * gate[base + tid];
}

__global__ __launch_bounds__(256) void add_kernel(float* __restrict__ x, const float* __restrict__ o)
{
  int i = blockIdx.x * 256 + threadIdx.x;
  if (i < TT * DD) x[i] += o[i];
}

// ---------------------------------------------------------------------------
// Host orchestration
// ---------------------------------------------------------------------------
struct LayerP {
  const float *ln_g,*ln_b,*ts,*Wr,*Wk,*Wv,*Wo,*d_base,*d_Wa,*d_Wb,
    *i_base,*i_Wa,*i_Wb,*dkm,*g_Wa,*g_Wb,*i_mix,*bonus,*gn_g,*gn_b,
    *cm_ln_g,*cm_ln_b,*cm_ts,*Wi,*Wo2,*v0_base,*v0_Wa,*v0_Wb;
};

extern "C" void kernel_launch(void* const* d_in, const int* in_sizes, int n_in,
                              void* d_out, int out_size, void* d_ws, size_t ws_size,
                              hipStream_t stream)
{
  (void)in_sizes; (void)n_in; (void)out_size; (void)ws_size;

  // setup_inputs() insertion order: tokens, then params depth-first
  const int* tokens = (const int*)d_in[0];
  int ci = 1;
  auto nf = [&]() { return (const float*)d_in[ci++]; };
  const float* embed = nf();
  const float* eln_g = nf();
  const float* eln_b = nf();
  LayerP Ls[2];
  for (int l = 0; l < 2; ++l) {
    LayerP& L = Ls[l];
    L.ln_g=nf(); L.ln_b=nf(); L.ts=nf(); L.Wr=nf(); L.Wk=nf(); L.Wv=nf(); L.Wo=nf();
    L.d_base=nf(); L.d_Wa=nf(); L.d_Wb=nf(); L.i_base=nf(); L.i_Wa=nf(); L.i_Wb=nf();
    L.dkm=nf(); L.g_Wa=nf(); L.g_Wb=nf(); L.i_mix=nf(); L.bonus=nf();
    L.gn_g=nf(); L.gn_b=nf(); L.cm_ln_g=nf(); L.cm_ln_b=nf(); L.cm_ts=nf();
    L.Wi=nf(); L.Wo2=nf();
    if (l > 0) { L.v0_base=nf(); L.v0_Wa=nf(); L.v0_Wb=nf(); }
    else       { L.v0_base=nullptr; L.v0_Wa=nullptr; L.v0_Wb=nullptr; }
  }
  const float* hln_g = nf();
  const float* hln_b = nf();
  const float* headW = nf();

  // workspace layout: 4MB [T,D] slabs
  const size_t S = (size_t)TT * DD;
  float* W = (float*)d_ws;
  float *x    = W +  0*S, *y    = W +  1*S;
  float *xr   = W +  2*S, *xd   = W +  3*S, *xk = W + 4*S;
  float *xv   = W +  5*S, *xi   = W +  6*S, *xg = W + 7*S;
  float *rb   = W +  8*S, *kb   = W +  9*S, *vb = W + 10*S;
  float *v0b  = W + 11*S, *dkb  = W + 12*S, *dec = W + 13*S;
  float *iclr = W + 14*S, *gate = W + 15*S, *sco = W + 16*S;
  float *mixo = W + 17*S, *tmpd = W + 18*S, *tmph = W + 19*S;
  float *h4   = W + 20*S;               // [T, 4096] uses 4 slabs

  const dim3 blk(256);
  const dim3 ge((TT * DD + 255) / 256);
  auto gemm = [&](const float* A, const float* B, float* C, int M, int N, int K,
                  int act, int ntst = 0) {
    dim3 grid((N + GTN - 1) / GTN, M / GTM);   // M % 128 == 0 at every call site
    gemm_bf16_wmma<<<grid, blk, 0, stream>>>(A, B, C, M, N, K, act, ntst);
  };

  embed_ln_kernel<<<TT, blk, 0, stream>>>(tokens, embed, eln_g, eln_b, x);

  for (int l = 0; l < 2; ++l) {
    const LayerP& L = Ls[l];
    // ---- time mixer ----
    ln_kernel<<<TT, blk, 0, stream>>>(x, L.ln_g, L.ln_b, y, 1e-5f);
    shift6_kernel<<<ge, blk, 0, stream>>>(y, L.ts, xr, xd, xk, xv, xi, xg);
    gemm(xr, L.Wr, rb, TT, DD, DD, 0);
    gemm(xk, L.Wk, kb, TT, DD, DD, 0);
    gemm(xv, L.Wv, vb, TT, DD, DD, 0);
    if (l == 0) {
      hipMemcpyAsync(v0b, vb, S * sizeof(float), hipMemcpyDeviceToDevice, stream);
    } else {
      gemm(xv, L.v0_Wa, tmph, TT, 32, DD, 0);
      gemm(tmph, L.v0_Wb, tmpd, TT, DD, 32, 0);
      vmix_kernel<<<ge, blk, 0, stream>>>(vb, v0b, tmpd, L.v0_base);
    }
    // gate = sigmoid(xg @ g_Wa) @ g_Wb
    gemm(xg, L.g_Wa, tmph, TT, 128, DD, 2);
    gemm(tmph, L.g_Wb, gate, TT, DD, 128, 0);
    // decay LoRA (tanh inner)
    gemm(xd, L.d_Wa, tmph, TT, 64, DD, 3);
    gemm(tmph, L.d_Wb, tmpd, TT, DD, 64, 0);
    decay_kernel<<<ge, blk, 0, stream>>>(tmpd, L.d_base, dec);
    // iclr LoRA (identity inner, sigmoid outer)
    gemm(xi, L.i_Wa, tmph, TT, 64, DD, 0);
    gemm(tmph, L.i_Wb, tmpd, TT, DD, 64, 0);
    sigb_kernel<<<ge, blk, 0, stream>>>(tmpd, L.i_base, iclr);
    // deformed key from pre-update k, then k-update
    dknorm_kernel<<<TT * HH, 64, 0, stream>>>(kb, L.dkm, dkb);
    kupd_kernel<<<ge, blk, 0, stream>>>(kb, iclr, L.i_mix);
    // sequential state scan
    scan_kernel<<<HH, 64, 0, stream>>>(rb, kb, vb, dec, iclr, dkb, sco);
    // groupnorm + bonus + gate, then output proj + residual
    postscan_kernel<<<TT * HH, 64, 0, stream>>>(sco, rb, kb, vb, gate,
                                                L.gn_g, L.gn_b, L.bonus, mixo);
    gemm(mixo, L.Wo, tmpd, TT, DD, DD, 0);
    add_kernel<<<ge, blk, 0, stream>>>(x, tmpd);

    // ---- channel mixer ----
    ln_kernel<<<TT, blk, 0, stream>>>(x, L.cm_ln_g, L.cm_ln_b, y, 1e-5f);
    shift1_kernel<<<ge, blk, 0, stream>>>(y, L.cm_ts, xr);
    gemm(xr, L.Wi, h4, TT, 4 * DD, DD, 1);       // relu^2 fused
    gemm(h4, L.Wo2, tmpd, TT, DD, 4 * DD, 0);
    add_kernel<<<ge, blk, 0, stream>>>(x, tmpd);
  }

  // final LN + head projection -> logits (fp32, write-once stream: NT stores)
  ln_kernel<<<TT, blk, 0, stream>>>(x, hln_g, hln_b, y, 1e-5f);
  gemm(y, headW, (float*)d_out, TT, 32000, DD, 0, 1);
}
